// AdaptiveInput_15556371546628
// MI455X (gfx1250) — matrix operand order repaired and checked by reading, not compiled, not run on hardware
//
#include <hip/hip_runtime.h>

typedef __attribute__((ext_vector_type(16))) __bf16 v16bf;
typedef __attribute__((ext_vector_type(8)))  float  v8f;

union FragBF { v16bf v; uint4 q[2]; };
union AccF   { v8f   v; float f[8]; };

#define NTOK  (8 * 4096)   // 32768 tokens
#define OUTD  1024

__device__ __forceinline__ unsigned short f2bf(float f) {
  // round-to-nearest-even f32 -> bf16
  unsigned u = __float_as_uint(f);
  u += 0x7FFFu + ((u >> 16) & 1u);
  return (unsigned short)(u >> 16);
}

// ---------------- phase 0: zero band counters ----------------
__global__ void zero_counts_kernel(int* counts) {
  if (threadIdx.x < 4) counts[threadIdx.x] = 0;
}

// ---------------- phase 1: classify tokens into bands ----------------
__global__ void classify_kernel(const int* __restrict__ tokens,
                                int* __restrict__ counts,
                                int* __restrict__ idxl, int n) {
  int i = blockIdx.x * 256 + threadIdx.x;
  if (i >= n) return;
  int tk = tokens[i];
  int band = (tk < 20000) ? 0 : ((tk < 60000) ? 1 : 2);
  int slot = atomicAdd(&counts[band], 1);
  idxl[band * NTOK + slot] = i;
}

// ---------------- phase 2: proj f32 -> bf16, pre-swizzled B fragments ----
// B fragment k order chosen here (contiguous, same as A): lane = n + 16*(kk/16), half = kk%16
// storage: Bsw[(((kt*64 + nt)*32 + lane)*16) + half]
__global__ void convertB_kernel(const float* __restrict__ proj,
                                unsigned short* __restrict__ Bsw,
                                int d, int total) {
  int e = blockIdx.x * 256 + threadIdx.x;
  if (e >= total) return;
  int half = e & 15;
  int lt   = e >> 4;
  int lane = lt & 31;
  int tile = lt >> 5;
  int nt   = tile & 63;
  int kt   = tile >> 6;
  int n    = lane & 15;
  int kk   = half + 16 * (lane >> 4);
  int k    = kt * 32 + kk;
  int o    = nt * 16 + n;
  Bsw[e] = f2bf(proj[(size_t)o * d + k]);
}

// ---------------- phase 3: per-band compacted GEMM ----------------
// WG = 256 threads = 8 waves. WG handles 32 compacted tokens (2 M-tiles) x N=1024.
// Each wave owns 8 N-tiles; acc = 2x8 v8f (128 VGPRs).
__global__ __launch_bounds__(256) void band_gemm_kernel(
    const float* __restrict__ emb, const unsigned short* __restrict__ Bsw,
    const int* __restrict__ idxl, const int* __restrict__ counts, int band,
    const int* __restrict__ tokens, float* __restrict__ out,
    int d, int KT, int band_start) {
  __shared__ unsigned short ldsA[2 * 16 * 32 * 16];  // 2 tiles x 16 kt x 512 halfs = 32 KB
  __shared__ int outpos[32];

  const int cnt = counts[band];
  const int slotBase = blockIdx.x * 32;
  if (slotBase >= cnt) return;  // uniform per block

  const int tid  = threadIdx.x;
  const int wave = tid >> 5;
  const int lane = tid & 31;
  const int ntBase = wave * 8;

  if (tid < 32) {
    int slot = slotBase + tid;
    outpos[tid] = (slot < cnt) ? idxl[slot] : -1;
  }

  v8f acc[2][8];
#pragma unroll
  for (int tt = 0; tt < 2; ++tt)
#pragma unroll
    for (int j = 0; j < 8; ++j) { v8f z = {}; acc[tt][j] = z; }

  for (int kt0 = 0; kt0 < KT; kt0 += 16) {
    const int ktn = (KT - kt0 < 16) ? (KT - kt0) : 16;
    __syncthreads();  // outpos ready (1st iter) / prev chunk consumed

    // stage A chunk: gather emb rows, cvt to bf16 pairs, pre-swizzled.
    // A frag k order chosen here (contiguous, same as B): lane = m + 16*(kk/16), half = kk%16
    const int pairsPerTok = ktn * 16;
    const int totalPairs  = 32 * pairsPerTok;
    for (int p = tid; p < totalPairs; p += 256) {
      const int mtok = p / pairsPerTok;
      const int kpl  = (p - mtok * pairsPerTok) * 2;  // even k within chunk
      const int tt = mtok >> 4, m = mtok & 15;
      unsigned packed = 0u;
      const int row = outpos[mtok];
      if (row >= 0) {
        const int token = tokens[row];
        const float* src = emb + (size_t)(token - band_start) * d + (kt0 * 32 + kpl);
        packed = (unsigned)f2bf(src[0]) | ((unsigned)f2bf(src[1]) << 16);
      }
      const int ktl   = kpl >> 5;
      const int kk    = kpl & 31;
      const int alane = m + 16 * (kk >> 4);
      const int ahalf = kk & 15;  // even
      *(unsigned*)&ldsA[(((tt * 16 + ktl) * 32 + alane) << 4) + ahalf] = packed;
    }
    __syncthreads();

    for (int ktl = 0; ktl < ktn; ++ktl) {
      FragBF a0, a1;
      const uint4* pa0 = (const uint4*)&ldsA[((0 * 16 + ktl) * 32 + lane) << 4];
      a0.q[0] = pa0[0]; a0.q[1] = pa0[1];
      const uint4* pa1 = (const uint4*)&ldsA[((1 * 16 + ktl) * 32 + lane) << 4];
      a1.q[0] = pa1[0]; a1.q[1] = pa1[1];
      const int kt = kt0 + ktl;
#pragma unroll
      for (int j = 0; j < 8; ++j) {
        FragBF b;
        const uint4* pb =
            (const uint4*)&Bsw[(((size_t)kt * 64 + (ntBase + j)) * 32 + lane) << 4];
        b.q[0] = pb[0]; b.q[1] = pb[1];
        acc[0][j] = __builtin_amdgcn_wmma_f32_16x16x32_bf16(
            false, a0.v, false, b.v, (short)0, acc[0][j], false, false);
        acc[1][j] = __builtin_amdgcn_wmma_f32_16x16x32_bf16(
            false, a1.v, false, b.v, (short)0, acc[1][j], false, false);
      }
    }
  }

  // C/D layout: VGPR r -> M = r + 8*(lane/16), N = lane%16. Scatter rows via idx list.
  const int n     = lane & 15;
  const int mBase = (lane >> 4) * 8;
#pragma unroll
  for (int tt = 0; tt < 2; ++tt) {
    int rowm[8];
#pragma unroll
    for (int r = 0; r < 8; ++r) rowm[r] = outpos[tt * 16 + mBase + r];
#pragma unroll
    for (int j = 0; j < 8; ++j) {
      AccF u; u.v = acc[tt][j];
      const int col = (ntBase + j) * 16 + n;
#pragma unroll
      for (int r = 0; r < 8; ++r)
        if (rowm[r] >= 0) out[(size_t)rowm[r] * OUTD + col] = u.f[r];
    }
  }
}

// ---------------- host launcher ----------------
extern "C" void kernel_launch(void* const* d_in, const int* in_sizes, int n_in,
                              void* d_out, int out_size, void* d_ws, size_t ws_size,
                              hipStream_t stream) {
  (void)in_sizes; (void)n_in; (void)out_size; (void)ws_size;
  const int*   tokens = (const int*)d_in[0];
  const float* emb0   = (const float*)d_in[1];
  const float* proj0  = (const float*)d_in[2];
  const float* emb1   = (const float*)d_in[3];
  const float* proj1  = (const float*)d_in[4];
  const float* emb2   = (const float*)d_in[5];
  const float* proj2  = (const float*)d_in[6];
  float* out = (float*)d_out;

  // workspace layout
  char* ws = (char*)d_ws;
  int* counts = (int*)ws;                                   // 16 B
  int* idxl   = (int*)(ws + 16);                            // 3 * 32768 * 4 B
  unsigned short* B0 = (unsigned short*)(ws + 393248);      // 32-aligned
  unsigned short* B1 = B0 + 32 * 64 * 512;                  // 1,048,576 halfs
  unsigned short* B2 = B1 + 8 * 64 * 512;                   //   262,144 halfs

  zero_counts_kernel<<<1, 32, 0, stream>>>(counts);
  classify_kernel<<<NTOK / 256, 256, 0, stream>>>(tokens, counts, idxl, NTOK);

  convertB_kernel<<<(32 * 64 * 512) / 256, 256, 0, stream>>>(proj0, B0, 1024, 32 * 64 * 512);
  convertB_kernel<<<( 8 * 64 * 512) / 256, 256, 0, stream>>>(proj1, B1,  256,  8 * 64 * 512);
  convertB_kernel<<<( 2 * 64 * 512) / 256, 256, 0, stream>>>(proj2, B2,   64,  2 * 64 * 512);

  // worst-case grids; blocks beyond ceil(count/32) exit immediately
  band_gemm_kernel<<<1024, 256, 0, stream>>>(emb0, B0, idxl,            counts, 0, tokens, out, 1024, 32,     0);
  band_gemm_kernel<<<1024, 256, 0, stream>>>(emb1, B1, idxl + NTOK,     counts, 1, tokens, out,  256,  8, 20000);
  band_gemm_kernel<<<1024, 256, 0, stream>>>(emb2, B2, idxl + 2 * NTOK, counts, 2, tokens, out,   64,  2, 60000);
}
